// MultiWaveletTransform_90366111908143
// MI455X (gfx1250) — matrix operand order repaired and checked by reading, not compiled, not run on hardware
//
#include <hip/hip_runtime.h>

// ---------------------------------------------------------------------------
// MultiWavelet transform for MI455X (gfx1250, wave32, WMMA).
// All dense contractions run on v_wmma_f32_16x16x32_f16 (f16 in, f32 acc).
// Big GEMMs use 16x64 wave tiles (A-fragment reused 4x) with pre-converted
// f16 activations so the K-loop is pure loads + WMMA.
// ---------------------------------------------------------------------------

typedef __attribute__((ext_vector_type(16))) _Float16 v16h;
typedef __attribute__((ext_vector_type(8)))  float    v8f;

#define TWO_PI_F 6.28318530717958647692f

__device__ __forceinline__ v8f wmma_f16(v16h a, v16h b, v8f c) {
  // (neg_a, A, neg_b, B, c_mod, C, reuse_a, reuse_b)
  return __builtin_amdgcn_wmma_f32_16x16x32_f16(false, a, false, b, (short)0, c,
                                                false, false);
}

// ---- fragment gathers per CDNA5 ISA 7.12.2 VGPR layouts (wave32) ----------

// A fragment (16x32, f16) from f16 row-major A[M][lda] at (row0, k0).
// Lane L: row = L%16; K-half = L/16 (+8); VGPR j holds K pairs.
__device__ __forceinline__ v16h load_a_frag_h(const _Float16* __restrict__ A,
                                              int lda, int row0, int k0) {
  int lane = threadIdx.x & 31;
  int r = lane & 15, h = lane >> 4;
  const _Float16* p = A + (size_t)(row0 + r) * lda + k0 + 8 * h;
  v16h a;
#pragma unroll
  for (int j = 0; j < 4; ++j) {
    a[2 * j]     = p[2 * j];      a[2 * j + 1]     = p[2 * j + 1];
    a[8 + 2 * j] = p[16 + 2 * j]; a[8 + 2 * j + 1] = p[16 + 2 * j + 1];
  }
  return a;
}

// A fragment from f16 row-major X[8][lda]; rows 8..15 are zero padding.
__device__ __forceinline__ v16h load_a_frag_h8(const _Float16* __restrict__ X,
                                               int lda, int k0) {
  int lane = threadIdx.x & 31;
  int r = lane & 15, h = lane >> 4;
  v16h a;
#pragma unroll
  for (int e = 0; e < 16; ++e) a[e] = (_Float16)0.f;
  if (r < 8) {
    const _Float16* p = X + (size_t)r * lda + k0 + 8 * h;
#pragma unroll
    for (int j = 0; j < 4; ++j) {
      a[2 * j]     = p[2 * j];      a[2 * j + 1]     = p[2 * j + 1];
      a[8 + 2 * j] = p[16 + 2 * j]; a[8 + 2 * j + 1] = p[16 + 2 * j + 1];
    }
  }
  return a;
}

// B fragment (32x16, f16) from N-major WT[N][ldb] (K contiguous) at (n0, k0).
// Lane L: col = L%16; K-half = L/16 (+8); VGPR j holds K pairs.
__device__ __forceinline__ v16h load_b_frag(const _Float16* __restrict__ WT,
                                            int ldb, int n0, int k0) {
  int lane = threadIdx.x & 31;
  int n = lane & 15, g = lane >> 4;
  const _Float16* p = WT + (size_t)(n0 + n) * ldb + k0 + 8 * g;
  v16h b;
#pragma unroll
  for (int j = 0; j < 8; ++j) { b[2 * j] = p[2 * j + (j >= 4 ? 8 : 0)]; b[2 * j + 1] = p[2 * j + 1 + (j >= 4 ? 8 : 0)]; }
  return b;
}

// ---------------------------------------------------------------------------
// Conversions / weight preparation
// ---------------------------------------------------------------------------

__global__ void k_f32_to_f16(const float* __restrict__ in,
                             _Float16* __restrict__ out, int count) {
  for (int idx = blockIdx.x * blockDim.x + threadIdx.x; idx < count;
       idx += gridDim.x * blockDim.x)
    out[idx] = (_Float16)in[idx];
}

// out[n*K + k] = (f16) in[k*N + n]   (f32 [K][N] -> f16 N-major [N][K])
__global__ void k_transpose_to_h(const float* __restrict__ in,
                                 _Float16* __restrict__ out, int K, int N) {
  int idx = blockIdx.x * blockDim.x + threadIdx.x;
  if (idx >= K * N) return;
  int n = idx / K, k = idx % K;
  out[idx] = (_Float16)in[(size_t)k * N + n];
}

// Mode weights: in (ck, ck, MODES) f32 -> out f16 [m][o][i] (N-major per mode)
__global__ void k_convert_mode_w(const float* __restrict__ in,
                                 _Float16* __restrict__ out, int CK, int MODES) {
  size_t total = (size_t)MODES * CK * CK;
  for (size_t idx = (size_t)blockIdx.x * blockDim.x + threadIdx.x; idx < total;
       idx += (size_t)gridDim.x * blockDim.x) {
    int i = (int)(idx % CK);
    size_t r = idx / CK;
    int o = (int)(r % CK);
    int m = (int)(r / CK);
    out[idx] = (_Float16)in[((size_t)i * CK + o) * MODES + m];
  }
}

// ---------------------------------------------------------------------------
// WMMA GEMM: C[M][N] = A[M][Kd] (f16) * W (f16, N-major [N][Kd]) + bias[N]
// One wave per 16x64 output tile: A fragment reused across 4 WMMAs per K-step.
// ---------------------------------------------------------------------------
__global__ void k_gemm_wmma_bias(const _Float16* __restrict__ A,
                                 const _Float16* __restrict__ WT,
                                 const float* __restrict__ bias,
                                 float* __restrict__ C, int M, int Kd, int N) {
  int wave = blockIdx.x * (blockDim.x >> 5) + (threadIdx.x >> 5);
  int tilesN = N >> 6;  // 64-wide wave tiles
  int mt = wave / tilesN, nt = wave % tilesN;
  if (mt * 16 >= M) return;
  int row0 = mt * 16, n0 = nt * 64;
  v8f acc[4];
#pragma unroll
  for (int t = 0; t < 4; ++t)
#pragma unroll
    for (int e = 0; e < 8; ++e) acc[t][e] = 0.f;
  for (int k0 = 0; k0 < Kd; k0 += 32) {
    v16h a = load_a_frag_h(A, Kd, row0, k0);
#pragma unroll
    for (int t = 0; t < 4; ++t) {
      v16h b = load_b_frag(WT, Kd, n0 + 16 * t, k0);
      acc[t] = wmma_f16(a, b, acc[t]);
    }
  }
  int lane = threadIdx.x & 31;
  int n = lane & 15, h = lane >> 4;
#pragma unroll
  for (int t = 0; t < 4; ++t) {
    int nc = n0 + 16 * t + n;
    float bv = bias ? bias[nc] : 0.f;
#pragma unroll
    for (int r = 0; r < 8; ++r) {
      int m = r + 8 * h;  // C/D layout: VGPR r holds row r (+8 for upper lanes)
      C[(size_t)(row0 + m) * N + nc] = acc[t][r] + bv;
    }
  }
}

// ---------------------------------------------------------------------------
// Complex mode-mixing GEMM:  Y[m] (8 x CK) = X[m] (8 x CK) * W[m] (CK x CK)
// re/im via 4 WMMAs per K-step.  One wave per (mode, 16-col tile) to keep
// modes*32 waves in flight (occupancy over reuse at this tiny M).
// ---------------------------------------------------------------------------
__global__ void k_mode_mix(const _Float16* __restrict__ Xr,
                           const _Float16* __restrict__ Xi,
                           const _Float16* __restrict__ Wr,
                           const _Float16* __restrict__ Wi,
                           float* __restrict__ Yr, float* __restrict__ Yi,
                           int modes, int CK, int accumulate) {
  int wave = blockIdx.x * (blockDim.x >> 5) + (threadIdx.x >> 5);
  int tilesN = CK >> 4;
  int m = wave / tilesN;
  int n0 = (wave % tilesN) << 4;
  if (m >= modes) return;
  const _Float16* xr = Xr + (size_t)m * 8 * CK;
  const _Float16* xi = Xi + (size_t)m * 8 * CK;
  const _Float16* wr = Wr + (size_t)m * CK * CK;
  const _Float16* wi = Wi + (size_t)m * CK * CK;
  int lane = threadIdx.x & 31;
  int n = lane & 15, h = lane >> 4;
  v8f ar, ai;
#pragma unroll
  for (int e = 0; e < 8; ++e) { ar[e] = 0.f; ai[e] = 0.f; }
  if (accumulate && h == 0) {
#pragma unroll
    for (int r = 0; r < 8; ++r) {
      size_t o = ((size_t)m * 8 + r) * CK + n0 + n;
      ar[r] = Yr[o]; ai[r] = Yi[o];
    }
  }
  for (int k0 = 0; k0 < CK; k0 += 32) {
    v16h fxr = load_a_frag_h8(xr, CK, k0);
    v16h fxi = load_a_frag_h8(xi, CK, k0);
    v16h fwr = load_b_frag(wr, CK, n0, k0);
    v16h fwi = load_b_frag(wi, CK, n0, k0);
    v16h fxin = -fxi;                       // FP16 WMMA has no A-negate modifier
    ar = wmma_f16(fxr, fwr, ar);
    ar = wmma_f16(fxin, fwi, ar);           // re -= xi*wi
    ai = wmma_f16(fxr, fwi, ai);
    ai = wmma_f16(fxi, fwr, ai);            // im += xi*wr
  }
  if (h == 0) {
#pragma unroll
    for (int r = 0; r < 8; ++r) {
      size_t o = ((size_t)m * 8 + r) * CK + n0 + n;
      Yr[o] = ar[r]; Yi[o] = ai[r];
    }
  }
}

// ---------------------------------------------------------------------------
// Wavelet decompose: x (B,2h,CK) -> s,d (B,h,CK) via ec_s/ec_d (16x8), LDS-staged
// ---------------------------------------------------------------------------
__global__ void k_decompose(const float* __restrict__ x,
                            const float* __restrict__ ec_s,
                            const float* __restrict__ ec_d,
                            float* __restrict__ s, float* __restrict__ d,
                            int half, int CK) {
  int j = blockIdx.x % half, b = blockIdx.x / half;
  __shared__ float xe[512], xo[512], fs[128], fd[128];
  int tid = threadIdx.x;
  const float* base = x + ((size_t)b * (2 * half) + 2 * j) * CK;
  xe[tid] = base[tid];
  xo[tid] = base[CK + tid];
  if (tid < 128) { fs[tid] = ec_s[tid]; fd[tid] = ec_d[tid]; }
  __syncthreads();
  int c = tid >> 3, ko = tid & 7;
  float as = 0.f, ad = 0.f;
#pragma unroll
  for (int kk = 0; kk < 8; ++kk) {
    float ve = xe[c * 8 + kk], vo = xo[c * 8 + kk];
    as += ve * fs[kk * 8 + ko] + vo * fs[(kk + 8) * 8 + ko];
    ad += ve * fd[kk * 8 + ko] + vo * fd[(kk + 8) * 8 + ko];
  }
  size_t o = ((size_t)b * half + j) * CK + tid;
  s[o] = as; d[o] = ad;
}

// ---------------------------------------------------------------------------
// Reconstruct: cat(x+Us, Ud) (B,m,CK->2K) * rc_e/rc_o, interleave -> (B,2m,CK)
// ---------------------------------------------------------------------------
__global__ void k_reconstruct(const float* __restrict__ x,
                              const float* __restrict__ Us,
                              const float* __restrict__ Ud,
                              const float* __restrict__ rc_e,
                              const float* __restrict__ rc_o,
                              float* __restrict__ xn, int m, int CK) {
  int j = blockIdx.x % m, b = blockIdx.x / m;
  __shared__ float lo[512], hi[512], fe[128], fo[128];
  int tid = threadIdx.x;
  size_t off = ((size_t)b * m + j) * CK;
  lo[tid] = x[off + tid] + Us[off + tid];
  hi[tid] = Ud[off + tid];
  if (tid < 128) { fe[tid] = rc_e[tid]; fo[tid] = rc_o[tid]; }
  __syncthreads();
  int c = tid >> 3, ko = tid & 7;
  float ae = 0.f, ao = 0.f;
#pragma unroll
  for (int kk = 0; kk < 8; ++kk) {
    float vl = lo[c * 8 + kk], vh = hi[c * 8 + kk];
    ae += vl * fe[kk * 8 + ko] + vh * fe[(kk + 8) * 8 + ko];
    ao += vl * fo[kk * 8 + ko] + vh * fo[(kk + 8) * 8 + ko];
  }
  size_t ob = ((size_t)b * (2 * m) + 2 * j) * CK + tid;
  xn[ob] = ae;
  xn[ob + CK] = ao;
}

// ---------------------------------------------------------------------------
// Truncated forward DFT (rfft, first `modes` bins): x (B,n,CK) f32 ->
// Xr/Xi f16 [m][b][ch].  grid = B*modes, block 256 (2 channels/thread).
// ---------------------------------------------------------------------------
__global__ void k_fwd_dft(const float* __restrict__ x, _Float16* __restrict__ Xr,
                          _Float16* __restrict__ Xi, int n, int modes, int CK) {
  int m = blockIdx.x % modes, b = blockIdx.x / modes;
  int ch = threadIdx.x;
  float w = -TWO_PI_F * (float)m / (float)n;
  float ar0 = 0.f, ai0 = 0.f, ar1 = 0.f, ai1 = 0.f;
  for (int t = 0; t < n; ++t) {
    float sv, cv;
    __sincosf(w * (float)t, &sv, &cv);
    const float* p = x + ((size_t)b * n + t) * CK;
    float v0 = p[ch], v1 = p[ch + 256];
    ar0 += v0 * cv; ai0 += v0 * sv;
    ar1 += v1 * cv; ai1 += v1 * sv;
  }
  size_t o = ((size_t)m * 8 + b) * CK + ch;
  Xr[o] = (_Float16)ar0;       Xi[o] = (_Float16)ai0;
  Xr[o + 256] = (_Float16)ar1; Xi[o + 256] = (_Float16)ai1;
}

// ---------------------------------------------------------------------------
// Truncated inverse rfft: Y [m][b][ch] f32 -> out (B,n,CK) f32.
// grid = B*n, block 256 (2 channels/thread).
// ---------------------------------------------------------------------------
__global__ void k_inv_dft(const float* __restrict__ Yr,
                          const float* __restrict__ Yi,
                          float* __restrict__ out, int n, int modes, int CK) {
  int t = blockIdx.x % n, b = blockIdx.x / n;
  int ch = threadIdx.x;
  float w = TWO_PI_F * (float)t / (float)n;
  float a0 = 0.f, a1 = 0.f;
  for (int m = 0; m < modes; ++m) {
    float sv, cv;
    __sincosf(w * (float)m, &sv, &cv);
    float coef = (m == 0 || 2 * m == n) ? 1.f : 2.f;  // DC / Nyquist counted once
    size_t o = ((size_t)m * 8 + b) * CK + ch;
    a0 += coef * (Yr[o] * cv - Yi[o] * sv);
    a1 += coef * (Yr[o + 256] * cv - Yi[o + 256] * sv);
  }
  float inv = 1.f / (float)n;
  size_t oo = ((size_t)b * n + t) * CK + ch;
  out[oo] = a0 * inv;
  out[oo + 256] = a1 * inv;
}

// Coarsest-scale k->k linear: y = x @ t0_w + t0_b, x (B,1,C,K)
__global__ void k_coarse_t0(const float* __restrict__ x,
                            const float* __restrict__ w,
                            const float* __restrict__ bias,
                            float* __restrict__ y, int CK) {
  int b = blockIdx.x, tid = threadIdx.x;
  int c = tid >> 3, ko = tid & 7;
  float acc = bias[ko];
#pragma unroll
  for (int kk = 0; kk < 8; ++kk) acc += x[(size_t)b * CK + c * 8 + kk] * w[kk * 8 + ko];
  y[(size_t)b * CK + tid] = acc;
}

// ---------------------------------------------------------------------------
// Host driver
// ---------------------------------------------------------------------------
extern "C" void kernel_launch(void* const* d_in, const int* in_sizes, int n_in,
                              void* d_out, int out_size, void* d_ws, size_t ws_size,
                              hipStream_t stream) {
  (void)in_sizes; (void)n_in; (void)out_size; (void)ws_size;
  const int B = 8, L = 1024, CK = 512, MODES = 16, NS = 10;
  const int M = B * L;  // 8192 GEMM rows

  const float* values = (const float*)d_in[2];
  const float* lk0_w = (const float*)d_in[3];
  const float* lk0_b = (const float*)d_in[4];
  const float* lk1_w = (const float*)d_in[5];
  const float* lk1_b = (const float*)d_in[6];
  const float* t0_w  = (const float*)d_in[7];
  const float* t0_b  = (const float*)d_in[8];
  const float* aw_re = (const float*)d_in[9];
  const float* aw_im = (const float*)d_in[10];
  const float* bw_re = (const float*)d_in[11];
  const float* bw_im = (const float*)d_in[12];
  const float* cw_re = (const float*)d_in[13];
  const float* cw_im = (const float*)d_in[14];
  const float* ec_s  = (const float*)d_in[15];
  const float* ec_d  = (const float*)d_in[16];
  const float* rc_e  = (const float*)d_in[17];
  const float* rc_o  = (const float*)d_in[18];

  // ---- workspace carve (all buffers written before read, every call) ----
  size_t off = 0;
  char* wsb = (char*)d_ws;
  auto alloc = [&](size_t bytes) -> void* {
    void* p = wsb + off;
    off += (bytes + 255) & ~(size_t)255;
    return p;
  };
  float* xA   = (float*)alloc((size_t)B * L * CK * 4);         // 16 MB
  float* xB   = (float*)alloc((size_t)B * L * CK * 4);         // 16 MB
  float* dbuf = (float*)alloc((size_t)B * (L / 2) * CK * 4);   // 8 MB
  float* Ud   = (float*)alloc((size_t)B * (L - 1) * CK * 4);   // stacks, 16.75 MB
  float* Us   = (float*)alloc((size_t)B * (L - 1) * CK * 4);
  _Float16* aHalf = (_Float16*)alloc((size_t)M * CK * 2);      // f16 GEMM A, 8 MB
  _Float16* xfr = (_Float16*)alloc((size_t)MODES * 8 * CK * 2);
  _Float16* xfi = (_Float16*)alloc((size_t)MODES * 8 * CK * 2);
  _Float16* dfr = (_Float16*)alloc((size_t)MODES * 8 * CK * 2);
  _Float16* dfi = (_Float16*)alloc((size_t)MODES * 8 * CK * 2);
  float* yUdr = (float*)alloc((size_t)MODES * 8 * CK * 4);
  float* yUdi = (float*)alloc((size_t)MODES * 8 * CK * 4);
  float* yUsr = (float*)alloc((size_t)MODES * 8 * CK * 4);
  float* yUsi = (float*)alloc((size_t)MODES * 8 * CK * 4);
  _Float16* lk0T = (_Float16*)alloc((size_t)CK * CK * 2);
  _Float16* lk1T = (_Float16*)alloc((size_t)CK * CK * 2);
  size_t mwBytes = (size_t)MODES * CK * CK * 2;                // 8 MB each
  _Float16* awTr = (_Float16*)alloc(mwBytes);
  _Float16* awTi = (_Float16*)alloc(mwBytes);
  _Float16* bwTr = (_Float16*)alloc(mwBytes);
  _Float16* bwTi = (_Float16*)alloc(mwBytes);
  _Float16* cwTr = (_Float16*)alloc(mwBytes);
  _Float16* cwTi = (_Float16*)alloc(mwBytes);

  // ---- weight prep (f32 -> f16, K-contiguous layouts for WMMA B operands) --
  {
    int tb = (CK * CK + 255) / 256;
    k_transpose_to_h<<<tb, 256, 0, stream>>>(lk0_w, lk0T, CK, CK);
    k_transpose_to_h<<<tb, 256, 0, stream>>>(lk1_w, lk1T, CK, CK);
    int gb = 8192;
    k_convert_mode_w<<<gb, 256, 0, stream>>>(aw_re, awTr, CK, MODES);
    k_convert_mode_w<<<gb, 256, 0, stream>>>(aw_im, awTi, CK, MODES);
    k_convert_mode_w<<<gb, 256, 0, stream>>>(bw_re, bwTr, CK, MODES);
    k_convert_mode_w<<<gb, 256, 0, stream>>>(bw_im, bwTi, CK, MODES);
    k_convert_mode_w<<<gb, 256, 0, stream>>>(cw_re, cwTr, CK, MODES);
    k_convert_mode_w<<<gb, 256, 0, stream>>>(cw_im, cwTi, CK, MODES);
  }

  const int nAct = M * CK;
  const int cvtBlocks = 4096;
  // 16x64 wave tiles: (M/16)*(N/64) waves, 8 waves per 256-thread block
  const int gemmBlocks = ((M / 16) * (CK / 64)) / 8;

  // ---- lk0: x = values @ lk0_w + b  -> xA (B,L,CK) ----
  k_f32_to_f16<<<cvtBlocks, 256, 0, stream>>>(values, aHalf, nAct);
  k_gemm_wmma_bias<<<gemmBlocks, 256, 0, stream>>>(aHalf, lk0T, lk0_b, xA, M, CK, CK);

  // ---- decomposition levels ----
  float* xin = xA;
  for (int i = 0; i < NS; ++i) {
    int half = 512 >> i;
    int modes = (half / 2 + 1 < MODES) ? (half / 2 + 1) : MODES;
    float* sOut = (i % 2 == 0) ? xB : xA;
    k_decompose<<<B * half, 512, 0, stream>>>(xin, ec_s, ec_d, sOut, dbuf, half, CK);
    k_fwd_dft<<<B * modes, 256, 0, stream>>>(dbuf, dfr, dfi, half, modes, CK);
    k_fwd_dft<<<B * modes, 256, 0, stream>>>(sOut, xfr, xfi, half, modes, CK);
    int mmBlocks = (modes * (CK / 16) + 7) / 8;
    k_mode_mix<<<mmBlocks, 256, 0, stream>>>(dfr, dfi, awTr, awTi, yUdr, yUdi, modes, CK, 0);
    k_mode_mix<<<mmBlocks, 256, 0, stream>>>(xfr, xfi, bwTr, bwTi, yUdr, yUdi, modes, CK, 1);
    k_mode_mix<<<mmBlocks, 256, 0, stream>>>(dfr, dfi, cwTr, cwTi, yUsr, yUsi, modes, CK, 0);
    size_t stackOff = (size_t)B * CK * (size_t)(1024 - 2 * half);
    k_inv_dft<<<B * half, 256, 0, stream>>>(yUdr, yUdi, Ud + stackOff, half, modes, CK);
    k_inv_dft<<<B * half, 256, 0, stream>>>(yUsr, yUsi, Us + stackOff, half, modes, CK);
    xin = sOut;
  }

  // ---- coarsest-scale linear (len 1) ----
  float* xcur = (xin == xA) ? xB : xA;
  k_coarse_t0<<<B, 512, 0, stream>>>(xin, t0_w, t0_b, xcur, CK);

  // ---- reconstruction levels ----
  for (int i = NS - 1; i >= 0; --i) {
    int m = 512 >> i;
    size_t stackOff = (size_t)B * CK * (size_t)(1024 - 2 * m);
    float* xn = (xcur == xA) ? xB : xA;
    k_reconstruct<<<B * m, 512, 0, stream>>>(xcur, Us + stackOff, Ud + stackOff,
                                             rc_e, rc_o, xn, m, CK);
    xcur = xn;
  }

  // ---- lk1: out = x @ lk1_w + b  -> d_out (B,L,H,D) == (B,L,512) flat ----
  k_f32_to_f16<<<cvtBlocks, 256, 0, stream>>>(xcur, aHalf, nAct);
  k_gemm_wmma_bias<<<gemmBlocks, 256, 0, stream>>>(aHalf, lk1T, lk1_b,
                                                   (float*)d_out, M, CK, CK);
}
